// HDCAM_65257733095815
// MI455X (gfx1250) — matrix operand order repaired and checked by reading, not compiled, not run on hardware
//
#include <hip/hip_runtime.h>

typedef __bf16 bf16_t;
typedef __attribute__((ext_vector_type(16))) __bf16 v16bf;
typedef __attribute__((ext_vector_type(8)))  float   v8f;

#define BB 8
#define TT 1024
#define DD 4096
#define SS 128
#define KK 8192
#define HH 512
#define VCB 256
#define MROWS (BB*TT)      /* 8192 token rows */
#define NAVIN (3*HH)       /* 1536 */

/* ---------------- WMMA GEMM: C[M,N] = act(scale * sum_k A[m,k]*B[n,k] + bias[n] + resid[m,n]) ----
 * 128x128 tile per 256-thread block (8 wave32 waves, 4x2 wave grid, 32x64 per wave),
 * double-buffered LDS pipeline: global_load_b128 for tile k+1 issues before the 8 WMMAs of
 * tile k, ds_store to the alternate buffer after, single barrier per K-step.             */
#define BM 128
#define BN 128
#define BK 32
#define PITCH 40           /* bf16 elems per LDS row: 80B, 16B aligned, conflict-free */

enum { ACT_NONE = 0, ACT_SIGMOID = 1, ACT_GELU = 2 };

union FragU { uint4 q[2]; v16bf v; };

__device__ inline v8f v8zero() {
  v8f v;
#pragma unroll
  for (int i = 0; i < 8; ++i) v[i] = 0.f;
  return v;
}

__global__ __launch_bounds__(256)
void wmma_gemm(const bf16_t* __restrict__ A, int lda,
               const bf16_t* __restrict__ Bm, int ldb,
               float* __restrict__ C32, int ldc32,
               bf16_t* __restrict__ C16, int ldc16,
               const float* __restrict__ bias,
               const float* __restrict__ resid, int ldres,
               float scale, int act, int Kd)
{
  __shared__ __align__(16) bf16_t As[2][BM * PITCH];
  __shared__ __align__(16) bf16_t Bs[2][BN * PITCH];

  const int tid  = threadIdx.x;
  const int wave = tid >> 5;
  const int lane = tid & 31;
  const int wm = wave & 3;        /* 4 waves tile M: 32 rows each  */
  const int wn = wave >> 2;       /* 2 waves tile N: 64 cols each  */
  const int gm0 = blockIdx.x * BM;
  const int gn0 = blockIdx.y * BN;
  const int lr = lane & 15, lh = lane >> 4;

  v8f acc[2][4];
#pragma unroll
  for (int mi = 0; mi < 2; ++mi)
#pragma unroll
    for (int ni = 0; ni < 4; ++ni) acc[mi][ni] = v8zero();

  /* staging: A 128x32 bf16 = 512 uint4, B 128x32 bf16 = 512 uint4, 2 each per thread */
  const int r0 = tid >> 2, r1 = (tid + 256) >> 2, seg = tid & 3;
  const bf16_t* Ap0 = &A[(size_t)(gm0 + r0) * lda + seg * 8];
  const bf16_t* Ap1 = &A[(size_t)(gm0 + r1) * lda + seg * 8];
  const bf16_t* Bp0 = &Bm[(size_t)(gn0 + r0) * ldb + seg * 8];
  const bf16_t* Bp1 = &Bm[(size_t)(gn0 + r1) * ldb + seg * 8];
  const int lofsA0 = r0 * PITCH + seg * 8, lofsA1 = r1 * PITCH + seg * 8;

  /* prologue: tile 0 -> buffer 0 */
  uint4 a0 = *(const uint4*)Ap0;
  uint4 a1 = *(const uint4*)Ap1;
  uint4 b0 = *(const uint4*)Bp0;
  uint4 b1 = *(const uint4*)Bp1;
  *(uint4*)&As[0][lofsA0] = a0;
  *(uint4*)&As[0][lofsA1] = a1;
  *(uint4*)&Bs[0][lofsA0] = b0;
  *(uint4*)&Bs[0][lofsA1] = b1;
  __syncthreads();

  int buf = 0;
  for (int k0 = 0; k0 < Kd; k0 += BK) {
    const bool more = (k0 + BK) < Kd;
    if (more) {           /* issue next tile's global loads before compute */
      a0 = *(const uint4*)(Ap0 + k0 + BK);
      a1 = *(const uint4*)(Ap1 + k0 + BK);
      b0 = *(const uint4*)(Bp0 + k0 + BK);
      b1 = *(const uint4*)(Bp1 + k0 + BK);
    }

    /* A fragment: ISA 16-bit A 16x32 layout (lanes 0-15: K0..7/K16..23; 16-31: K8..15/K24..31) */
    v16bf af[2], bfr[4];
#pragma unroll
    for (int mi = 0; mi < 2; ++mi) {
      FragU f;
      int r = wm * 32 + mi * 16 + lr;
      f.q[0] = *(const uint4*)&As[buf][r * PITCH + lh * 8];
      f.q[1] = *(const uint4*)&As[buf][r * PITCH + 16 + lh * 8];
      af[mi] = f.v;
    }
    /* B fragment: same K layout as A (lanes 0-15: K0..7/K16..23; 16-31: K8..15/K24..31); N = lane%16 */
#pragma unroll
    for (int ni = 0; ni < 4; ++ni) {
      FragU f;
      int r = wn * 64 + ni * 16 + lr;
      f.q[0] = *(const uint4*)&Bs[buf][r * PITCH + lh * 8];
      f.q[1] = *(const uint4*)&Bs[buf][r * PITCH + 16 + lh * 8];
      bfr[ni] = f.v;
    }
#pragma unroll
    for (int mi = 0; mi < 2; ++mi)
#pragma unroll
      for (int ni = 0; ni < 4; ++ni)
        acc[mi][ni] = __builtin_amdgcn_wmma_f32_16x16x32_bf16(
            false, af[mi], false, bfr[ni], (short)0, acc[mi][ni], false, false);

    if (more) {           /* fill the alternate buffer */
      *(uint4*)&As[buf ^ 1][lofsA0] = a0;
      *(uint4*)&As[buf ^ 1][lofsA1] = a1;
      *(uint4*)&Bs[buf ^ 1][lofsA0] = b0;
      *(uint4*)&Bs[buf ^ 1][lofsA1] = b1;
    }
    __syncthreads();
    buf ^= 1;
  }

  /* epilogue: C layout lanes0-15 N=lane M=r; lanes16-31 N=lane-16 M=8+r */
#pragma unroll
  for (int mi = 0; mi < 2; ++mi)
#pragma unroll
    for (int ni = 0; ni < 4; ++ni) {
      int row0 = gm0 + wm * 32 + mi * 16 + lh * 8;
      int col  = gn0 + wn * 64 + ni * 16 + lr;
      float bc = bias ? bias[col] : 0.f;
#pragma unroll
      for (int r = 0; r < 8; ++r) {
        int row = row0 + r;
        float v = acc[mi][ni][r] * scale + bc;
        if (resid) v += resid[(size_t)row * ldres + col];
        if (act == ACT_SIGMOID)      v = 1.f / (1.f + __expf(-v));
        else if (act == ACT_GELU)    v = 0.5f * v * (1.f + erff(v * 0.70710678118f));
        if (C32) C32[(size_t)row * ldc32 + col] = v;
        if (C16) C16[(size_t)row * ldc16 + col] = (bf16_t)v;
      }
    }
}

/* ---------------- trigram encode: x = 0.7*c0[d]*c1[d-1]*c2[d-2] + 0.3*c0[d] ---------------- */
__global__ __launch_bounds__(256)
void trigram_kernel(const int* __restrict__ idx, const float* __restrict__ cb,
                    float* __restrict__ xf, bf16_t* __restrict__ xb)
{
  int row = blockIdx.x;            /* b*T + t */
  int t = row & (TT - 1);
  const float* c0 = cb + (size_t)idx[row] * DD;
  const float* c1 = (t >= 1) ? cb + (size_t)idx[row - 1] * DD : nullptr;
  const float* c2 = (t >= 2) ? cb + (size_t)idx[row - 2] * DD : nullptr;
  for (int d = threadIdx.x; d < DD; d += 256) {
    float a = c0[d];
    float m1 = c1 ? c1[(d - 1) & (DD - 1)] : 0.f;
    float m2 = c2 ? c2[(d - 2) & (DD - 1)] : 0.f;
    float v = 0.7f * a * m1 * m2 + 0.3f * a;
    xf[(size_t)row * DD + d] = v;
    xb[(size_t)row * DD + d] = (bf16_t)v;
  }
}

/* ---------------- GLA sequential scan: state[s,u] = g[s]*state + k[s]*v[u]; out[u]=sum_s q[s]*state ---- */
__global__ __launch_bounds__(512)
void gla_scan(const float* __restrict__ qf, const float* __restrict__ kf,
              const float* __restrict__ vf, const float* __restrict__ gf,
              float* __restrict__ out)
{
  int b = blockIdx.x;
  int u = threadIdx.x;             /* 0..127 state column    */
  int sq = threadIdx.y;            /* 0..3  s-quarter        */
  __shared__ float qs[SS], ks[SS], vs[SS], gs[SS];
  __shared__ float part[4][SS];
  float st[32];
#pragma unroll
  for (int i = 0; i < 32; ++i) st[i] = 0.f;

  for (int t = 0; t < TT; ++t) {
    size_t base = ((size_t)b * TT + t) * SS;
    int tl = sq * 128 + u;
    if (tl < 128)      qs[tl]       = qf[base + tl];
    else if (tl < 256) ks[tl - 128] = kf[base + tl - 128];
    else if (tl < 384) vs[tl - 256] = vf[base + tl - 256];
    else               gs[tl - 384] = gf[base + tl - 384];
    __syncthreads();
    float vv = vs[u];
    float acc = 0.f;
    int s0 = sq * 32;
#pragma unroll
    for (int i = 0; i < 32; ++i) {
      st[i] = gs[s0 + i] * st[i] + ks[s0 + i] * vv;
      acc += qs[s0 + i] * st[i];
    }
    part[sq][u] = acc;
    __syncthreads();
    if (sq == 0) out[base + u] = part[0][u] + part[1][u] + part[2][u] + part[3][u];
  }
}

/* ---------------- LayerNorm over width W, bf16 output ---------------- */
__global__ __launch_bounds__(256)
void ln_kernel(const float* __restrict__ in, int ld, int W,
               const float* __restrict__ g, const float* __restrict__ b,
               bf16_t* __restrict__ out, int ldo)
{
  int row = blockIdx.x, tid = threadIdx.x;
  const float* x = in + (size_t)row * ld;
  __shared__ float red[256];
  float s = 0.f;
  for (int i = tid; i < W; i += 256) s += x[i];
  red[tid] = s; __syncthreads();
  for (int o = 128; o; o >>= 1) { if (tid < o) red[tid] += red[tid + o]; __syncthreads(); }
  float mean = red[0] / W; __syncthreads();
  float v = 0.f;
  for (int i = tid; i < W; i += 256) { float d = x[i] - mean; v += d * d; }
  red[tid] = v; __syncthreads();
  for (int o = 128; o; o >>= 1) { if (tid < o) red[tid] += red[tid + o]; __syncthreads(); }
  float rstd = rsqrtf(red[0] / W + 1e-5f);
  for (int i = tid; i < W; i += 256)
    out[(size_t)row * ldo + i] = (bf16_t)((x[i] - mean) * rstd * g[i] + b[i]);
}

/* ---------------- row softmax -> bf16 probabilities ---------------- */
__global__ __launch_bounds__(256)
void softmax_kernel(const float* __restrict__ S, bf16_t* __restrict__ P, int N)
{
  int row = blockIdx.x, tid = threadIdx.x;
  const float* x = S + (size_t)row * N;
  __shared__ float red[256];
  float m = -3.4e38f;
  for (int i = tid; i < N; i += 256) m = fmaxf(m, x[i]);
  red[tid] = m; __syncthreads();
  for (int o = 128; o; o >>= 1) { if (tid < o) red[tid] = fmaxf(red[tid], red[tid + o]); __syncthreads(); }
  m = red[0]; __syncthreads();
  float s = 0.f;
  for (int i = tid; i < N; i += 256) s += __expf(x[i] - m);
  red[tid] = s; __syncthreads();
  for (int o = 128; o; o >>= 1) { if (tid < o) red[tid] += red[tid + o]; __syncthreads(); }
  float inv = 1.f / red[0];
  for (int i = tid; i < N; i += 256)
    P[(size_t)row * N + i] = (bf16_t)(__expf(x[i] - m) * inv);
}

/* ---------------- residual stream: res_embed[idx] + pos_embed[t] ---------------- */
__global__ __launch_bounds__(256)
void resid_kernel(const int* __restrict__ idx, const float* __restrict__ re,
                  const float* __restrict__ pe, float* __restrict__ dst, int ldd)
{
  int row = blockIdx.x;
  int t = row & (TT - 1);
  const float* r0 = re + (size_t)idx[row] * HH;
  const float* p0 = pe + (size_t)t * HH;
  for (int i = threadIdx.x; i < HH; i += 256)
    dst[(size_t)row * ldd + i] = r0[i] + p0[i];
}

/* ---------------- FiLM: hmod = h*(1+gamma)+beta, bf16 out ---------------- */
__global__ __launch_bounds__(256)
void film_kernel(const float* __restrict__ h, const float* __restrict__ film,
                 bf16_t* __restrict__ outb)
{
  int row = blockIdx.x;
  for (int i = threadIdx.x; i < HH; i += 256) {
    float g = film[(size_t)row * (2 * HH) + i];
    float b = film[(size_t)row * (2 * HH) + HH + i];
    outb[(size_t)row * HH + i] = (bf16_t)(h[(size_t)row * HH + i] * (1.f + g) + b);
  }
}

/* ---------------- weight prep: f32 (R,C) -> bf16 (C,R); and plain convert ---------------- */
__global__ __launch_bounds__(256)
void transpose_bf16(const float* __restrict__ in, bf16_t* __restrict__ out, int R, int C)
{
  size_t n = (size_t)R * C;
  for (size_t i = (size_t)blockIdx.x * blockDim.x + threadIdx.x; i < n;
       i += (size_t)gridDim.x * blockDim.x) {
    size_t r = i / C, c = i % C;
    out[c * (size_t)R + r] = (bf16_t)in[i];
  }
}

__global__ __launch_bounds__(256)
void convert_bf16(const float* __restrict__ in, bf16_t* __restrict__ out, size_t n)
{
  for (size_t i = (size_t)blockIdx.x * blockDim.x + threadIdx.x; i < n;
       i += (size_t)gridDim.x * blockDim.x)
    out[i] = (bf16_t)in[i];
}

/* =================================================================================== */
extern "C" void kernel_launch(void* const* d_in, const int* in_sizes, int n_in,
                              void* d_out, int out_size, void* d_ws, size_t ws_size,
                              hipStream_t stream)
{
  const int*   idx      = (const int*)d_in[0];
  const float* cb       = (const float*)d_in[1];
  const float* Wq = (const float*)d_in[2];  const float* bq = (const float*)d_in[3];
  const float* Wk = (const float*)d_in[4];  const float* bk = (const float*)d_in[5];
  const float* Wv = (const float*)d_in[6];  const float* bv = (const float*)d_in[7];
  const float* Wg = (const float*)d_in[8];  const float* bg = (const float*)d_in[9];
  const float* Wo = (const float*)d_in[10]; const float* bo = (const float*)d_in[11];
  const float* gln_g = (const float*)d_in[12]; const float* gln_b = (const float*)d_in[13];
  const float* mkeys = (const float*)d_in[14]; const float* mvals = (const float*)d_in[15];
  const float* Whdc  = (const float*)d_in[16]; const float* bhdc  = (const float*)d_in[17];
  const float* re    = (const float*)d_in[18]; const float* pe    = (const float*)d_in[19];
  const float* nln_g = (const float*)d_in[20]; const float* nln_b = (const float*)d_in[21];
  const float* W1 = (const float*)d_in[22]; const float* b1 = (const float*)d_in[23];
  const float* W2 = (const float*)d_in[24]; const float* b2 = (const float*)d_in[25];
  const float* W3 = (const float*)d_in[26]; const float* b3 = (const float*)d_in[27];
  const float* Wf = (const float*)d_in[28]; const float* bfilm = (const float*)d_in[29];
  const float* Wh = (const float*)d_in[30]; const float* bh = (const float*)d_in[31];
  float* out = (float*)d_out;

  char* ws = (char*)d_ws;
  size_t off = 0;
  auto alloc = [&](size_t bytes) -> char* {
    char* p = ws + off;
    off += (bytes + 255) & ~(size_t)255;
    return p;
  };

  float*  xf    = (float*)alloc((size_t)MROWS * DD * 4);   /* also pre-LN hdc (in-place resid) */
  bf16_t* xb    = (bf16_t*)alloc((size_t)MROWS * DD * 2);
  float*  qf    = (float*)alloc((size_t)MROWS * SS * 4);
  float*  kf    = (float*)alloc((size_t)MROWS * SS * 4);
  float*  vf    = (float*)alloc((size_t)MROWS * SS * 4);
  float*  gf    = (float*)alloc((size_t)MROWS * SS * 4);
  float*  glaf  = (float*)alloc((size_t)MROWS * SS * 4);
  bf16_t* glab  = (bf16_t*)alloc((size_t)MROWS * SS * 2);
  bf16_t* hdcb  = (bf16_t*)alloc((size_t)MROWS * DD * 2);
  bf16_t* keysb = (bf16_t*)alloc((size_t)KK * DD * 2);
  bf16_t* valT  = (bf16_t*)alloc((size_t)HH * KK * 2);
  bf16_t* WqT   = (bf16_t*)alloc((size_t)SS * DD * 2);
  bf16_t* WkT   = (bf16_t*)alloc((size_t)SS * DD * 2);
  bf16_t* WvT   = (bf16_t*)alloc((size_t)SS * DD * 2);
  bf16_t* WgT   = (bf16_t*)alloc((size_t)SS * DD * 2);
  bf16_t* WoT   = (bf16_t*)alloc((size_t)DD * SS * 2);
  bf16_t* WhdcT = (bf16_t*)alloc((size_t)HH * DD * 2);
  bf16_t* W1T   = (bf16_t*)alloc((size_t)HH * NAVIN * 2);
  bf16_t* W2T   = (bf16_t*)alloc((size_t)HH * HH * 2);
  bf16_t* W3T   = (bf16_t*)alloc((size_t)HH * HH * 2);
  bf16_t* WfT   = (bf16_t*)alloc((size_t)(2 * HH) * HH * 2);
  bf16_t* WhT   = (bf16_t*)alloc((size_t)VCB * HH * 2);
  float*  scores = (float*)alloc((size_t)MROWS * KK * 4);
  bf16_t* attnb  = (bf16_t*)alloc((size_t)MROWS * KK * 2);
  float*  comb   = (float*)alloc((size_t)MROWS * NAVIN * 4);
  bf16_t* combln = (bf16_t*)alloc((size_t)MROWS * NAVIN * 2);
  bf16_t* hdcFb  = (bf16_t*)alloc((size_t)MROWS * HH * 2);
  bf16_t* h1b    = (bf16_t*)alloc((size_t)MROWS * HH * 2);
  bf16_t* h2b    = (bf16_t*)alloc((size_t)MROWS * HH * 2);
  float*  h3f    = (float*)alloc((size_t)MROWS * HH * 4);
  float*  filmf  = (float*)alloc((size_t)MROWS * 2 * HH * 4);
  bf16_t* hmodb  = (bf16_t*)alloc((size_t)MROWS * HH * 2);
  (void)ws_size; (void)n_in; (void)in_sizes; (void)out_size;

  /* ---- weight prep: transpose+convert to bf16 (N,K) layout ---- */
  transpose_bf16<<<2048, 256, 0, stream>>>(Wq, WqT, DD, SS);
  transpose_bf16<<<2048, 256, 0, stream>>>(Wk, WkT, DD, SS);
  transpose_bf16<<<2048, 256, 0, stream>>>(Wv, WvT, DD, SS);
  transpose_bf16<<<2048, 256, 0, stream>>>(Wg, WgT, DD, SS);
  transpose_bf16<<<2048, 256, 0, stream>>>(Wo, WoT, SS, DD);
  transpose_bf16<<<2048, 256, 0, stream>>>(Whdc, WhdcT, DD, HH);
  transpose_bf16<<<2048, 256, 0, stream>>>(W1, W1T, NAVIN, HH);
  transpose_bf16<<<1024, 256, 0, stream>>>(W2, W2T, HH, HH);
  transpose_bf16<<<1024, 256, 0, stream>>>(W3, W3T, HH, HH);
  transpose_bf16<<<1024, 256, 0, stream>>>(Wf, WfT, HH, 2 * HH);
  transpose_bf16<<<1024, 256, 0, stream>>>(Wh, WhT, HH, VCB);
  transpose_bf16<<<2048, 256, 0, stream>>>(mvals, valT, KK, HH);
  convert_bf16<<<4096, 256, 0, stream>>>(mkeys, keysb, (size_t)KK * DD);

  auto GEMM = [&](const bf16_t* A, int lda, const bf16_t* Bm, int ldb, int N, int Kd,
                  float* C32, int ldc32, bf16_t* C16, int ldc16,
                  const float* bias, const float* resid, int ldres,
                  float scale, int act) {
    dim3 g(MROWS / BM, N / BN);
    wmma_gemm<<<g, 256, 0, stream>>>(A, lda, Bm, ldb, C32, ldc32, C16, ldc16,
                                     bias, resid, ldres, scale, act, Kd);
  };

  /* ---- encode trigrams ---- */
  trigram_kernel<<<MROWS, 256, 0, stream>>>(idx, cb, xf, xb);

  /* ---- GLA projections (M=8192, K=4096, N=128) ---- */
  GEMM(xb, DD, WqT, DD, SS, DD, qf, SS, nullptr, 0, bq, nullptr, 0, 1.f, ACT_NONE);
  GEMM(xb, DD, WkT, DD, SS, DD, kf, SS, nullptr, 0, bk, nullptr, 0, 1.f, ACT_NONE);
  GEMM(xb, DD, WvT, DD, SS, DD, vf, SS, nullptr, 0, bv, nullptr, 0, 1.f, ACT_NONE);
  GEMM(xb, DD, WgT, DD, SS, DD, gf, SS, nullptr, 0, bg, nullptr, 0, 1.f, ACT_SIGMOID);

  /* ---- sequential gated scan ---- */
  gla_scan<<<dim3(BB), dim3(128, 4, 1), 0, stream>>>(qf, kf, vf, gf, glaf);
  convert_bf16<<<2048, 256, 0, stream>>>(glaf, glab, (size_t)MROWS * SS);

  /* ---- out proj fused with residual add (in-place into xf), then LN ---- */
  GEMM(glab, SS, WoT, SS, DD, SS, xf, DD, nullptr, 0, bo, xf, DD, 1.f, ACT_NONE);
  ln_kernel<<<MROWS, 256, 0, stream>>>(xf, DD, DD, gln_g, gln_b, hdcb, DD);

  /* ---- associative memory: scores -> softmax -> retrieved ---- */
  GEMM(hdcb, DD, keysb, DD, KK, DD, scores, KK, nullptr, 0, nullptr, nullptr, 0,
       0.015625f /* 4096^-0.5 */, ACT_NONE);
  softmax_kernel<<<MROWS, 256, 0, stream>>>(scores, attnb, KK);
  GEMM(attnb, KK, valT, KK, HH, KK, comb + HH, NAVIN, nullptr, 0,
       nullptr, nullptr, 0, 1.f, ACT_NONE);

  /* ---- combined streams ---- */
  GEMM(hdcb, DD, WhdcT, DD, HH, DD, comb, NAVIN, hdcFb, HH, bhdc, nullptr, 0, 1.f, ACT_NONE);
  resid_kernel<<<MROWS, 256, 0, stream>>>(idx, re, pe, comb + 2 * HH, NAVIN);
  ln_kernel<<<MROWS, 256, 0, stream>>>(comb, NAVIN, NAVIN, nln_g, nln_b, combln, NAVIN);

  /* ---- navigator MLP + FiLM + head ---- */
  GEMM(combln, NAVIN, W1T, NAVIN, HH, NAVIN, nullptr, 0, h1b, HH, b1, nullptr, 0, 1.f, ACT_GELU);
  GEMM(h1b, HH, W2T, HH, HH, HH, nullptr, 0, h2b, HH, b2, nullptr, 0, 1.f, ACT_GELU);
  GEMM(h2b, HH, W3T, HH, HH, HH, h3f, HH, nullptr, 0, b3, nullptr, 0, 1.f, ACT_GELU);
  GEMM(hdcFb, HH, WfT, HH, 2 * HH, HH, filmf, 2 * HH, nullptr, 0, bfilm, nullptr, 0, 1.f, ACT_NONE);
  film_kernel<<<MROWS, 256, 0, stream>>>(h3f, filmf, hmodb);
  GEMM(hmodb, HH, WhT, HH, VCB, HH, out, VCB, nullptr, 0, bh, nullptr, 0, 1.f, ACT_NONE);
}
